// LightweightConv1dTBC_60138132078898
// MI455X (gfx1250) — matrix operand order repaired and checked by reading, not compiled, not run on hardware
//
#include <hip/hip_runtime.h>

// ---- WMMA types (CDNA5 / gfx1250, wave32) ----
typedef __attribute__((ext_vector_type(16))) __bf16 v16bf;
typedef __attribute__((ext_vector_type(8)))  __bf16 v8bf;
typedef __attribute__((ext_vector_type(8)))  float  v8f;

#define T_DIM   2048
#define B_DIM   32
#define C_DIM   1024
#define H_DIM   16
#define K_TAPS  31
#define PAD_T   15
#define R_DIM   (C_DIM / H_DIM)   // 64 channels per head

#define TT      512               // output times per workgroup
#define WINDOW  560               // staged input times: TT + 48 (halo + K=64 band pad)
#define PITCH   568               // LDS time pitch (bf16): 284 dwords, stride%64=28 -> conflict-free b128

__global__ __launch_bounds__(256)
void lconv_tbc_wmma(const float* __restrict__ x,
                    const float* __restrict__ weight,
                    const float* __restrict__ bias,
                    float* __restrict__ y) {
    // [channel][time] bf16 window; padded pitch dodges bank conflicts on b128 reads
    __shared__ __bf16 lds_x[16 * PITCH];
    __shared__ float  wsm[32];                 // softmaxed kernel taps

    const int c0     = blockIdx.x * 16;        // 16-channel group (within one head)
    const int b      = blockIdx.y;
    const int t_base = blockIdx.z * TT;
    const int head   = c0 / R_DIM;
    const int tid    = threadIdx.x;

    // ---- softmax over 31 taps (tiny; one thread, serial) ----
    if (tid == 0) {
        float wv[K_TAPS];
        float m = -1e30f;
        #pragma unroll
        for (int j = 0; j < K_TAPS; ++j) { wv[j] = weight[head * K_TAPS + j]; m = fmaxf(m, wv[j]); }
        float s = 0.f;
        #pragma unroll
        for (int j = 0; j < K_TAPS; ++j) { wv[j] = __expf(wv[j] - m); s += wv[j]; }
        const float inv = 1.f / s;
        #pragma unroll
        for (int j = 0; j < K_TAPS; ++j) wsm[j] = wv[j] * inv;
        wsm[31] = 0.f;
    }

    // ---- stage x window -> LDS (transpose to [ch][time], fp32 -> bf16) ----
    // One float4 = 4 channels at one time (16B aligned; 4 lanes cover a 64B row).
    for (int idx = tid; idx < WINDOW * 4; idx += 256) {
        const int t_rel = idx >> 2;
        const int q4    = (idx & 3) * 4;
        const int t_in  = t_base - PAD_T + t_rel;
        float4 v = make_float4(0.f, 0.f, 0.f, 0.f);
        if (t_in >= 0 && t_in < T_DIM)
            v = *(const float4*)&x[((size_t)t_in * B_DIM + b) * C_DIM + c0 + q4];
        // zero-padded halo: banded-A zeros then multiply finite zeros (no 0*NaN)
        lds_x[(q4 + 0) * PITCH + t_rel] = (__bf16)v.x;
        lds_x[(q4 + 1) * PITCH + t_rel] = (__bf16)v.y;
        lds_x[(q4 + 2) * PITCH + t_rel] = (__bf16)v.z;
        lds_x[(q4 + 3) * PITCH + t_rel] = (__bf16)v.w;
    }
    __syncthreads();

    const int lane = tid & 31;
    const int wave = tid >> 5;
    const int half = lane >> 4;     // K-half selector per ISA 16-bit layouts
    const int mn   = lane & 15;     // M for A-frag, N for B/C/D-frags

    // ---- A fragments (banded softmax weights), 16-bit A 16x32, contiguous k order ----
    // lane element e -> K = e + 16*half
    v16bf a0, a1;
    #pragma unroll
    for (int e = 0; e < 16; ++e) {
        const int K    = e + 16 * half;
        const int tap0 = K - mn;            // chunk 0: band k = 0..31
        const int tap1 = K + 32 - mn;       // chunk 1: band k = 32..63
        a0[e] = (__bf16)((tap0 >= 0 && tap0 < K_TAPS) ? wsm[tap0] : 0.f);
        a1[e] = (__bf16)((tap1 >= 0 && tap1 < K_TAPS) ? wsm[tap1] : 0.f);
    }

    const float bv = bias[c0 + mn];

    // ---- 4 output tiles per wave; 2 WMMAs per tile (K = 64 band) ----
    #pragma unroll
    for (int i = wave * 4; i < wave * 4 + 4; ++i) {
        const int toff = 16 * i;  // window-relative base: (t0-15) - (t_base-15)

        // B frag: N = channel (lane), K = time. 16 consecutive bf16, 16B aligned
        // -> two ds_load_b128 per chunk. Here lanes 16-31 hold K+16 (contiguous order, as A).
        const __bf16* p0 = &lds_x[mn * PITCH + toff + 16 * half];
        const __bf16* p1 = &lds_x[mn * PITCH + toff + 32 + 16 * half];
        union { v16bf v; v8bf h[2]; } b0u, b1u;
        b0u.h[0] = *(const v8bf*)(p0);  b0u.h[1] = *(const v8bf*)(p0 + 8);
        b1u.h[0] = *(const v8bf*)(p1);  b1u.h[1] = *(const v8bf*)(p1 + 8);

        v8f acc = {};
        acc = __builtin_amdgcn_wmma_f32_16x16x32_bf16(false, a0, false, b0u.v,
                                                      (short)0, acc, false, false);
        acc = __builtin_amdgcn_wmma_f32_16x16x32_bf16(false, a1, false, b1u.v,
                                                      (short)0, acc, false, false);

        // D layout: lane n = l&15; VGPR v -> output row M = v + 8*half.
        const int t0 = t_base + 16 * i;
        #pragma unroll
        for (int v = 0; v < 8; ++v) {
            const int t_out = t0 + v + 8 * half;
            y[((size_t)t_out * B_DIM + b) * C_DIM + c0 + mn] = acc[v] + bv;
        }
    }
}

extern "C" void kernel_launch(void* const* d_in, const int* in_sizes, int n_in,
                              void* d_out, int out_size, void* d_ws, size_t ws_size,
                              hipStream_t stream) {
    (void)in_sizes; (void)n_in; (void)d_ws; (void)ws_size; (void)out_size;
    const float* x      = (const float*)d_in[0];
    const float* weight = (const float*)d_in[1];
    const float* bias   = (const float*)d_in[2];
    float*       y      = (float*)d_out;

    dim3 grid(C_DIM / 16, B_DIM, T_DIM / TT);   // (64, 32, 4)
    dim3 block(256);                             // 8 wave32 waves
    lconv_tbc_wmma<<<grid, block, 0, stream>>>(x, weight, bias, y);
}
